// QuantizedMoE_10488310137395
// MI455X (gfx1250) — matrix operand order repaired and checked by reading, not compiled, not run on hardware
//
#include <hip/hip_runtime.h>

// ---------------- problem constants ----------------
#define DD 1024      // model dim
#define FF 2048      // ffn dim
#define TT 4096      // tokens (B*S)
#define NE 8         // routed experts
#define E9 9         // routed + shared
#define TILE_M 32    // tokens per block tile (2 WMMA M-tiles per wave)
#define FC 128       // F-chunk width

typedef __attribute__((ext_vector_type(16))) __bf16 v16bf;
typedef __attribute__((ext_vector_type(8)))  float  v8f;

union FragB16 { v16bf v; uint4 q[2]; };

__device__ __forceinline__ unsigned short f32_to_bf16(float f) {
    unsigned u = __float_as_uint(f);
    unsigned r = 0x7FFFu + ((u >> 16) & 1u);
    return (unsigned short)((u + r) >> 16);
}

__device__ __forceinline__ void atomic_add_f32(float* p, float v) {
    __hip_atomic_fetch_add(p, v, __ATOMIC_RELAXED, __HIP_MEMORY_SCOPE_AGENT);
}

// ---------------- kernel 1: init out + counters ----------------
__global__ void qmoe_init_kernel(float* __restrict__ out, int n, int* __restrict__ counts) {
    for (int i = blockIdx.x * blockDim.x + threadIdx.x; i < n; i += gridDim.x * blockDim.x)
        out[i] = 0.0f;
    if (blockIdx.x == 0 && threadIdx.x < E9)
        counts[threadIdx.x] = (threadIdx.x == NE) ? TT : 0;
}

// ---------------- kernel 2: f32 -> bf16 (routed || shared concatenated) ----------------
__global__ void qmoe_cvt_kernel(const float* __restrict__ a, long na,
                                const float* __restrict__ b, long nb,
                                unsigned short* __restrict__ dst) {
    long n = na + nb;
    for (long i = blockIdx.x * (long)blockDim.x + threadIdx.x; i < n;
         i += (long)gridDim.x * blockDim.x) {
        float v = (i < na) ? a[i] : b[i - na];
        dst[i] = f32_to_bf16(v);
    }
}

// ---------------- kernel 3: router + dispatch ----------------
__global__ __launch_bounds__(256) void qmoe_router_kernel(
    const float* __restrict__ x, const float* __restrict__ rw, const float* __restrict__ rb,
    int* __restrict__ tok_list, float* __restrict__ gate_list, int* __restrict__ counts) {
    const int wave = threadIdx.x >> 5;
    const int lane = threadIdx.x & 31;
    const int t = blockIdx.x * 8 + wave;

    float acc[NE];
#pragma unroll
    for (int e = 0; e < NE; ++e) acc[e] = 0.0f;
    const float* xr = x + (size_t)t * DD;
    for (int i = lane; i < DD; i += 32) {
        float xv = xr[i];
#pragma unroll
        for (int e = 0; e < NE; ++e) acc[e] += xv * rw[e * DD + i];
    }
#pragma unroll
    for (int off = 16; off > 0; off >>= 1) {
#pragma unroll
        for (int e = 0; e < NE; ++e) acc[e] += __shfl_xor(acc[e], off, 32);
    }
    if (lane == 0) {
        float mx = -3.4e38f;
#pragma unroll
        for (int e = 0; e < NE; ++e) { acc[e] += rb[e]; mx = fmaxf(mx, acc[e]); }
        float p[NE];
#pragma unroll
        for (int e = 0; e < NE; ++e) p[e] = __expf(acc[e] - mx);
        int i0 = 0; float v0 = p[0];
#pragma unroll
        for (int e = 1; e < NE; ++e) if (p[e] > v0) { v0 = p[e]; i0 = e; }
        int i1 = -1; float v1 = -1.0f;
#pragma unroll
        for (int e = 0; e < NE; ++e) if (e != i0 && p[e] > v1) { v1 = p[e]; i1 = e; }
        float s = v0 + v1;
        int s0 = atomicAdd(&counts[i0], 1);
        tok_list[i0 * TT + s0] = t; gate_list[i0 * TT + s0] = v0 / s;
        int s1 = atomicAdd(&counts[i1], 1);
        tok_list[i1 * TT + s1] = t; gate_list[i1 * TT + s1] = v1 / s;
        // shared expert: identity slot, gate 1
        tok_list[NE * TT + t] = t; gate_list[NE * TT + t] = 1.0f;
    }
}

// ---------------- kernel 4: fused up->silu->down per expert tile (WMMA bf16) ----------------
__global__ __launch_bounds__(256) void qmoe_moe_kernel(
    const unsigned short* __restrict__ xb,
    const unsigned short* __restrict__ wub,   // [E9][FF][DD] bf16
    const unsigned short* __restrict__ wdb,   // [E9][DD][FF] bf16
    const int* __restrict__ tok_list, const float* __restrict__ gate_list,
    const int* __restrict__ counts, float* __restrict__ out) {

    __shared__ unsigned short xs[TILE_M][DD];   // 64 KB staged X rows (bf16)
    __shared__ unsigned short hs[TILE_M][FC];   // 8 KB silu(h) chunk (bf16)
    __shared__ int   s_tok[TILE_M];
    __shared__ float s_gate[TILE_M];

    const int e    = blockIdx.y;
    const int tile = blockIdx.x;
    const int count = counts[e];
    if (tile * TILE_M >= count) return;

    const int tid  = threadIdx.x;
    const int w    = tid >> 5;     // wave 0..7
    const int lane = tid & 31;
    const int ln   = lane & 15;
    const int hi   = lane >> 4;    // half-wave select

    if (tid < TILE_M) {
        int a = tile * TILE_M + tid;
        if (a < count) { s_tok[tid] = tok_list[e * TT + a]; s_gate[tid] = gate_list[e * TT + a]; }
        else           { s_tok[tid] = 0;                    s_gate[tid] = 0.0f; }
    }
    __syncthreads();

    // stage 32 gathered rows of x (bf16) into LDS: 4096 uint4, 16 per thread
    {
        uint4* xs4 = reinterpret_cast<uint4*>(&xs[0][0]);
#pragma unroll
        for (int i = 0; i < 16; ++i) {
            int idx = tid + 256 * i;        // 0..4095
            int r   = idx >> 7;             // 128 uint4 per row
            int c   = idx & 127;
            const uint4* src = reinterpret_cast<const uint4*>(xb + (size_t)s_tok[r] * DD + c * 8);
            xs4[idx] = *src;
        }
    }
    __syncthreads();

    const size_t wu_base = (size_t)e * FF * DD;
    const size_t wd_base = (size_t)e * DD * FF;

    v8f zero = {};
    v8f acc_dn0[8], acc_dn1[8];
#pragma unroll
    for (int i = 0; i < 8; ++i) { acc_dn0[i] = zero; acc_dn1[i] = zero; }

    for (int chunk = 0; chunk < FF / FC; ++chunk) {
        const int f_base = chunk * FC;

        // ---- up-projection: this wave's two 16x16 h tiles at columns f_base + w*16
        v8f acc_up0 = zero, acc_up1 = zero;
        const int f_col = f_base + w * 16;
        for (int ks = 0; ks < DD / 32; ++ks) {
            const int k = ks * 32;
            FragB16 a0, a1, b;
            // A (16x32 bf16): lane<16 row=ln K{0..7,16..23}; lane>=16 K{8..15,24..31}
            a0.q[0] = *reinterpret_cast<const uint4*>(&xs[ln][k + hi * 8]);
            a0.q[1] = *reinterpret_cast<const uint4*>(&xs[ln][k + 16 + hi * 8]);
            a1.q[0] = *reinterpret_cast<const uint4*>(&xs[16 + ln][k + hi * 8]);
            a1.q[1] = *reinterpret_cast<const uint4*>(&xs[16 + ln][k + 16 + hi * 8]);
            // B (32x16 bf16): lane<16 col=ln K{0..7,16..23}; lane>=16 K{8..15,24..31} (w_up row contiguous in D)
            const uint4* pb = reinterpret_cast<const uint4*>(
                wub + wu_base + (size_t)(f_col + ln) * DD + k + hi * 8);
            b.q[0] = pb[0]; b.q[1] = pb[2];
            acc_up0 = __builtin_amdgcn_wmma_f32_16x16x32_bf16(
                false, a0.v, false, b.v, (short)0, acc_up0, false, false);
            acc_up1 = __builtin_amdgcn_wmma_f32_16x16x32_bf16(
                false, a1.v, false, b.v, (short)0, acc_up1, false, false);
        }

        __syncthreads();   // prior chunk's down-phase reads of hs are done
        // SiLU + write h tiles to LDS per C layout (lane<16: M=r; lane>=16: M=r+8)
#pragma unroll
        for (int r = 0; r < 8; ++r) {
            float h0 = acc_up0[r];
            float s0 = h0 / (1.0f + __expf(-h0));
            hs[r + 8 * hi][w * 16 + ln] = f32_to_bf16(s0);
            float h1 = acc_up1[r];
            float s1 = h1 / (1.0f + __expf(-h1));
            hs[16 + r + 8 * hi][w * 16 + ln] = f32_to_bf16(s1);
        }
        __syncthreads();

        // ---- down-projection: wave owns output dims [w*128, w*128+128), K = FC
        for (int ks = 0; ks < FC / 32; ++ks) {
            const int k = ks * 32;
            FragB16 a0, a1;
            a0.q[0] = *reinterpret_cast<const uint4*>(&hs[ln][k + hi * 8]);
            a0.q[1] = *reinterpret_cast<const uint4*>(&hs[ln][k + 16 + hi * 8]);
            a1.q[0] = *reinterpret_cast<const uint4*>(&hs[16 + ln][k + hi * 8]);
            a1.q[1] = *reinterpret_cast<const uint4*>(&hs[16 + ln][k + 16 + hi * 8]);
#pragma unroll
            for (int nt = 0; nt < 8; ++nt) {
                const int n_col = w * 128 + nt * 16;
                FragB16 b;
                const uint4* pb = reinterpret_cast<const uint4*>(
                    wdb + wd_base + (size_t)(n_col + ln) * FF + f_base + k + hi * 8);
                b.q[0] = pb[0]; b.q[1] = pb[2];
                acc_dn0[nt] = __builtin_amdgcn_wmma_f32_16x16x32_bf16(
                    false, a0.v, false, b.v, (short)0, acc_dn0[nt], false, false);
                acc_dn1[nt] = __builtin_amdgcn_wmma_f32_16x16x32_bf16(
                    false, a1.v, false, b.v, (short)0, acc_dn1[nt], false, false);
            }
        }
    }

    // ---- gate-scaled accumulation into out (token appears in <=3 lists -> atomics)
#pragma unroll
    for (int nt = 0; nt < 8; ++nt) {
        const int n = w * 128 + nt * 16 + ln;
#pragma unroll
        for (int r = 0; r < 8; ++r) {
            const int m0 = r + 8 * hi;
            atomic_add_f32(out + (size_t)s_tok[m0] * DD + n, acc_dn0[nt][r] * s_gate[m0]);
            const int m1 = 16 + r + 8 * hi;
            atomic_add_f32(out + (size_t)s_tok[m1] * DD + n, acc_dn1[nt][r] * s_gate[m1]);
        }
    }
}

// ---------------- launch ----------------
extern "C" void kernel_launch(void* const* d_in, const int* in_sizes, int n_in,
                              void* d_out, int out_size, void* d_ws, size_t ws_size,
                              hipStream_t stream) {
    const float* x       = (const float*)d_in[0];
    const float* rw      = (const float*)d_in[1];
    const float* rb      = (const float*)d_in[2];
    const float* w_up    = (const float*)d_in[3];
    const float* w_down  = (const float*)d_in[4];
    const float* sw_up   = (const float*)d_in[5];
    const float* sw_down = (const float*)d_in[6];
    float* out = (float*)d_out;

    // workspace layout
    char* ws = (char*)d_ws;
    const size_t sz_xb  = (size_t)TT * DD * 2;           // 8 MB
    const size_t sz_w   = (size_t)E9 * FF * DD * 2;      // 37.75 MB each
    const size_t sz_tok = (size_t)E9 * TT * 4;
    size_t off = 0;
    unsigned short* xb  = (unsigned short*)(ws + off); off += sz_xb;
    unsigned short* wub = (unsigned short*)(ws + off); off += sz_w;
    unsigned short* wdb = (unsigned short*)(ws + off); off += sz_w;
    int*   tok_list  = (int*)(ws + off);   off += sz_tok;
    float* gate_list = (float*)(ws + off); off += sz_tok;
    int*   counts    = (int*)(ws + off);   off += 256;

    qmoe_init_kernel<<<1024, 256, 0, stream>>>(out, TT * DD, counts);

    qmoe_cvt_kernel<<<4096, 256, 0, stream>>>(w_up,   (long)NE * FF * DD, sw_up,   (long)FF * DD, wub);
    qmoe_cvt_kernel<<<4096, 256, 0, stream>>>(w_down, (long)NE * DD * FF, sw_down, (long)DD * FF, wdb);
    qmoe_cvt_kernel<<<2048, 256, 0, stream>>>(x, (long)TT * DD, x, 0, xb);

    qmoe_router_kernel<<<TT / 8, 256, 0, stream>>>(x, rw, rb, tok_list, gate_list, counts);

    dim3 grid(TT / TILE_M, E9);
    qmoe_moe_kernel<<<grid, 256, 0, stream>>>(xb, wub, wdb, tok_list, gate_list, counts, out);
}
